// DDGNN_81733227643386
// MI455X (gfx1250) — compile-verified, matrix operand order checked by reading, not run on hardware
//
#include <hip/hip_runtime.h>
#include <hip/hip_bf16.h>

// ---------------------------------------------------------------- constants
constexpr int kB = 16, kT = 64, kN = 1024, kD = 32, kH = 64, kC = 32, kO = 12;
constexpr float kAlpha = 0.1f;
constexpr int kKprop = 10;
constexpr int kNT = kN / 64;               // 16 tiles of 64 along N
constexpr int kPairs = kNT * (kNT + 1) / 2; // 136 upper-triangular tile pairs

typedef __attribute__((ext_vector_type(16))) _Float16 v16h;
typedef __attribute__((ext_vector_type(8)))  _Float16 v8h;
typedef __attribute__((ext_vector_type(8)))  float    v8f;
typedef __attribute__((ext_vector_type(2)))  float    v2f;

// Load a 16-f16 A/B fragment for v_wmma_f32_16x16x32_f16.
// Per-lane K map: e in [0,8) -> K = ks + e + 8*half ; e in [8,16) -> K = ks + 16 + (e-8) + 8*half
__device__ inline v16h load_frag16(const _Float16* p, int half) {
  v8h t0 = *(const v8h*)(p + 8 * half);
  v8h t1 = *(const v8h*)(p + 16 + 8 * half);
  v16h a;
#pragma unroll
  for (int e = 0; e < 8; ++e) { a[e] = t0[e]; a[8 + e] = t1[e]; }
  return a;
}

__device__ inline float sigmoidf_(float v) { return 1.0f / (1.0f + expf(-v)); }

// ============================================================ 1) embeddings
// m1 = xl@w1+b1, m2 = xl@w2+b2 stored as f16 row-major (b*N, H) for WMMA.
__global__ void embed_kernel(const float* __restrict__ x,
                             const float* __restrict__ w1, const float* __restrict__ b1,
                             const float* __restrict__ w2, const float* __restrict__ b2,
                             _Float16* __restrict__ m1h, _Float16* __restrict__ m2h) {
  __shared__ float w1s[kD * kH], w2s[kD * kH], b1s[kH], b2s[kH];
  int tid = threadIdx.x;
  for (int i = tid; i < kD * kH; i += 256) { w1s[i] = w1[i]; w2s[i] = w2[i]; }
  if (tid < kH) { b1s[tid] = b1[tid]; b2s[tid] = b2[tid]; }
  __syncthreads();
  int p = blockIdx.x * 256 + tid;            // 0..B*N-1
  int b = p >> 10, n = p & (kN - 1);
  const float* xr = x + (((size_t)b * kT + (kT - 1)) * kN + n) * kD;
  float xv[kD];
#pragma unroll
  for (int d = 0; d < kD; ++d) xv[d] = xr[d];
  for (int hh = 0; hh < kH; ++hh) {
    float a1 = b1s[hh], a2 = b2s[hh];
#pragma unroll
    for (int d = 0; d < kD; ++d) { a1 += xv[d] * w1s[d * kH + hh]; a2 += xv[d] * w2s[d * kH + hh]; }
    m1h[(size_t)p * kH + hh] = (_Float16)a1;
    m2h[(size_t)p * kH + hh] = (_Float16)a2;
  }
}

// =========================================== 2) adjacency GEMM + symmetrize
// For tile pair (i,j), i<=j: S1 = M1_i M2_j^T, S2 = M1_j M2_i^T (f16 WMMA, K=64),
// write tanh(S1 + S2^T) to block (i,j) and its transpose to (j,i).
__global__ void adjgemm_kernel(const _Float16* __restrict__ m1h,
                               const _Float16* __restrict__ m2h,
                               float* __restrict__ adjout) {
  int b = blockIdx.x / kPairs;
  int rem = blockIdx.x % kPairs;
  int i = 0;
  while (rem >= kNT - i) { rem -= kNT - i; ++i; }
  int j = i + rem;

  int tid = threadIdx.x, wave = tid >> 5, lane = tid & 31;
  int half = lane >> 4, lr = lane & 15;
  const _Float16* M1 = m1h + (size_t)b * kN * kH;
  const _Float16* M2 = m2h + (size_t)b * kN * kH;

  v8f acc1[4] = {}, acc2[4] = {};
  int n1 = i * 64 + wave * 16 + lr;   // rows of S1
  int n2 = j * 64 + wave * 16 + lr;   // rows of S2
#pragma unroll
  for (int ks = 0; ks < kH; ks += 32) {
    v16h a1 = load_frag16(M1 + (size_t)n1 * kH + ks, half);
    v16h a2 = load_frag16(M1 + (size_t)n2 * kH + ks, half);
#pragma unroll
    for (int ct = 0; ct < 4; ++ct) {
      v16h bf1 = load_frag16(M2 + (size_t)(j * 64 + ct * 16 + lr) * kH + ks, half);
      v16h bf2 = load_frag16(M2 + (size_t)(i * 64 + ct * 16 + lr) * kH + ks, half);
      acc1[ct] = __builtin_amdgcn_wmma_f32_16x16x32_f16(false, a1, false, bf1,
                                                        (short)0, acc1[ct], false, false);
      acc2[ct] = __builtin_amdgcn_wmma_f32_16x16x32_f16(false, a2, false, bf2,
                                                        (short)0, acc2[ct], false, false);
    }
  }

  __shared__ float S1[64 * 64], S2[64 * 64];
#pragma unroll
  for (int ct = 0; ct < 4; ++ct)
#pragma unroll
    for (int r = 0; r < 8; ++r) {
      int rl = wave * 16 + r + 8 * half;  // C/D: M = r + 8*half
      int cl = ct * 16 + lr;              // C/D: N = lane%16
      S1[rl * 64 + cl] = acc1[ct][r];
      S2[rl * 64 + cl] = acc2[ct][r];
    }
  __syncthreads();

  float* adjb = adjout + (size_t)b * kN * kN;
  for (int e = tid; e < 4096; e += 128) {          // block (i,j), coalesced in m
    int nl = e >> 6, ml = e & 63;
    float t = tanhf(S1[nl * 64 + ml] + S2[ml * 64 + nl]);
    adjb[(size_t)(i * 64 + nl) * kN + (j * 64 + ml)] = t;
  }
  if (i != j) {
    for (int e = tid; e < 4096; e += 128) {        // block (j,i), coalesced in n
      int ml = e >> 6, nl = e & 63;
      float t = tanhf(S1[nl * 64 + ml] + S2[ml * 64 + nl]);
      adjb[(size_t)(j * 64 + ml) * kN + (i * 64 + nl)] = t;
    }
  }
}

// ============================================= 3) row softmax + D^-1/2 terms
__global__ void softmax_kernel(float* __restrict__ adj, float* __restrict__ dis) {
  float* r = adj + (size_t)blockIdx.x * kN;
  int tid = threadIdx.x;
  __shared__ float red[256];
  float v[4];
  float lmax = -1e30f;
#pragma unroll
  for (int i = 0; i < 4; ++i) { v[i] = r[tid + i * 256]; lmax = fmaxf(lmax, v[i]); }
  red[tid] = lmax; __syncthreads();
  for (int s = 128; s > 0; s >>= 1) { if (tid < s) red[tid] = fmaxf(red[tid], red[tid + s]); __syncthreads(); }
  float m = red[0]; __syncthreads();
  float lsum = 0.f;
#pragma unroll
  for (int i = 0; i < 4; ++i) { v[i] = expf(v[i] - m); lsum += v[i]; }
  red[tid] = lsum; __syncthreads();
  for (int s = 128; s > 0; s >>= 1) { if (tid < s) red[tid] += red[tid + s]; __syncthreads(); }
  float inv = 1.0f / red[0]; __syncthreads();
  float psum = 0.f;
#pragma unroll
  for (int i = 0; i < 4; ++i) { float u = v[i] * inv; r[tid + i * 256] = u; psum += u; }
  red[tid] = psum; __syncthreads();
  for (int s = 128; s > 0; s >>= 1) { if (tid < s) red[tid] += red[tid + s]; __syncthreads(); }
  if (tid == 0) dis[blockIdx.x] = rsqrtf(red[0] + 1.0f);   // deg = rowsum + 1
}

// ================================== 4) pruned causal gated convs (layer 1)
// Only t=49..63 of x feed the final timestep. Layer-1 outputs needed at
// u = 51+2*ti (ti=0..6); input time index rel. 49 is tr = 2*ti + k.
__global__ void conv1_kernel(const float* __restrict__ x,
                             const float* __restrict__ fw, const float* __restrict__ fb,
                             const float* __restrict__ gw, const float* __restrict__ gb,
                             float* __restrict__ act1) {
  __shared__ float fws[kC * kD * 3], gws[kC * kD * 3], fbs[kC], gbs[kC];
  __shared__ float xs[8][15 * kD];
  int tid = threadIdx.x;
  for (int idx = tid; idx < kC * kD * 3; idx += 256) { fws[idx] = fw[idx]; gws[idx] = gw[idx]; }
  if (tid < kC) { fbs[tid] = fb[tid]; gbs[tid] = gb[tid]; }
  int g = tid >> 5, lane = tid & 31;
  int p = blockIdx.x * 8 + g;                // (b,n) flat
  int b = p >> 10, n = p & (kN - 1);
  const float* xb = x + (((size_t)b * kT + 49) * kN + n) * kD;
#pragma unroll
  for (int tr = 0; tr < 15; ++tr) xs[g][tr * kD + lane] = xb[(size_t)tr * kN * kD + lane];
  __syncthreads();
  int c = lane;
  for (int ti = 0; ti < 7; ++ti) {
    float f = fbs[c], gg = gbs[c];
#pragma unroll
    for (int k = 0; k < 3; ++k) {
      const float* xr = &xs[g][(2 * ti + k) * kD];
#pragma unroll
      for (int d = 0; d < kD; ++d) {
        float xv = xr[d];
        f  += fws[c * 96 + d * 3 + k] * xv;
        gg += gws[c * 96 + d * 3 + k] * xv;
      }
    }
    act1[(size_t)p * 224 + ti * kC + c] = tanhf(f) * sigmoidf_(gg);
  }
}

// ============================ 5) layers 2 (dil=2) + 3 (dil=4) fused -> h
__global__ void conv23_kernel(const float* __restrict__ act1,
                              const float* __restrict__ fw2, const float* __restrict__ fb2,
                              const float* __restrict__ gw2, const float* __restrict__ gb2,
                              const float* __restrict__ fw3, const float* __restrict__ fb3,
                              const float* __restrict__ gw3, const float* __restrict__ gb3,
                              float* __restrict__ h) {
  __shared__ float fw2s[3072], gw2s[3072], fw3s[3072], gw3s[3072];
  __shared__ float fb2s[kC], gb2s[kC], fb3s[kC], gb3s[kC];
  __shared__ float a1s[8][224], a2s[8][96];
  int tid = threadIdx.x;
  for (int idx = tid; idx < 3072; idx += 256) {
    fw2s[idx] = fw2[idx]; gw2s[idx] = gw2[idx];
    fw3s[idx] = fw3[idx]; gw3s[idx] = gw3[idx];
  }
  if (tid < kC) { fb2s[tid] = fb2[tid]; gb2s[tid] = gb2[tid]; fb3s[tid] = fb3[tid]; gb3s[tid] = gb3[tid]; }
  int g = tid >> 5, lane = tid & 31;
  int p = blockIdx.x * 8 + g;
#pragma unroll
  for (int i = lane; i < 224; i += 32) a1s[g][i] = act1[(size_t)p * 224 + i];
  __syncthreads();
  int c = lane;
#pragma unroll
  for (int j = 0; j < 3; ++j) {               // layer-2 at u = 55+4j, uses a1 idx 2j+k
    float f = fb2s[c], gg = gb2s[c];
#pragma unroll
    for (int k = 0; k < 3; ++k) {
      const float* ar = &a1s[g][(2 * j + k) * kC];
#pragma unroll
      for (int d = 0; d < kC; ++d) {
        float av = ar[d];
        f  += fw2s[c * 96 + d * 3 + k] * av;
        gg += gw2s[c * 96 + d * 3 + k] * av;
      }
    }
    a2s[g][j * kC + c] = tanhf(f) * sigmoidf_(gg);
  }
  __syncthreads();
  {                                            // layer-3 at t=63 uses a2 idx k
    float f = fb3s[c], gg = gb3s[c];
#pragma unroll
    for (int k = 0; k < 3; ++k) {
      const float* ar = &a2s[g][k * kC];
#pragma unroll
      for (int d = 0; d < kC; ++d) {
        float av = ar[d];
        f  += fw3s[c * 96 + d * 3 + k] * av;
        gg += gw3s[c * 96 + d * 3 + k] * av;
      }
    }
    h[(size_t)p * kC + c] = tanhf(f) * sigmoidf_(gg);
  }
}

// ================== 6) propagation sweep: zout = a*h + (1-a)*Dn(adj+I)Dn zin
// fp32 WMMA 16x16x4; D^-1/2 scaling + identity folded into A-fragment load;
// 64x32 z panel staged in LDS per K-chunk (L2-resident adj, LDS-resident z).
__global__ void prop_kernel(const float* __restrict__ adj, const float* __restrict__ dis,
                            const float* __restrict__ h, const float* __restrict__ zin,
                            float* __restrict__ zout) {
  int b = blockIdx.x >> 4;
  int nblk = blockIdx.x & 15;
  int tid = threadIdx.x, wave = tid >> 5, lane = tid & 31;
  int half = lane >> 4, lr = lane & 15;
  __shared__ float zs[64 * kC];
  __shared__ float diss[kN];
  for (int i = tid; i < kN; i += 128) diss[i] = dis[b * kN + i];
  __syncthreads();
  int nstrip = nblk * 64 + wave * 16;
  int n_row = nstrip + lr;                         // A: M = lane%16
  float disn = diss[n_row];
  const float* adjr = adj + ((size_t)b * kN + n_row) * kN;
  const float* zb = zin + (size_t)b * kN * kC;
  v8f acc0 = {}, acc1 = {};
  for (int mc = 0; mc < 16; ++mc) {
    __syncthreads();
    for (int e = tid; e < 64 * kC; e += 128) zs[e] = zb[(size_t)mc * 64 * kC + e];
    __syncthreads();
#pragma unroll
    for (int ks = 0; ks < 64; ks += 4) {
      int mx = mc * 64 + ks + 2 * half;            // A: K = 2*half + r
      float ax = adjr[mx], ay = adjr[mx + 1];
      ax = (ax + (n_row == mx     ? 1.0f : 0.0f)) * (disn * diss[mx]);
      ay = (ay + (n_row == mx + 1 ? 1.0f : 0.0f)) * (disn * diss[mx + 1]);
      v2f a; a[0] = ax; a[1] = ay;
      int ml = ks + 2 * half;
      v2f b0, b1;
      b0[0] = zs[ml * kC + lr];       b0[1] = zs[(ml + 1) * kC + lr];
      b1[0] = zs[ml * kC + 16 + lr];  b1[1] = zs[(ml + 1) * kC + 16 + lr];
      acc0 = __builtin_amdgcn_wmma_f32_16x16x4_f32(false, a, false, b0, (short)0, acc0, false, false);
      acc1 = __builtin_amdgcn_wmma_f32_16x16x4_f32(false, a, false, b1, (short)0, acc1, false, false);
    }
  }
  const float* hb = h + (size_t)b * kN * kC;
  float* zo = zout + (size_t)b * kN * kC;
#pragma unroll
  for (int r = 0; r < 8; ++r) {                    // D: M = r + 8*half, N = lane%16
    int n = nstrip + r + 8 * half;
    zo[(size_t)n * kC + lr]      = kAlpha * hb[(size_t)n * kC + lr]      + (1.0f - kAlpha) * acc0[r];
    zo[(size_t)n * kC + 16 + lr] = kAlpha * hb[(size_t)n * kC + 16 + lr] + (1.0f - kAlpha) * acc1[r];
  }
}

// ============================================ 7) out = relu(z) @ wo + bo
__global__ void out_kernel(const float* __restrict__ z, const float* __restrict__ wo,
                           const float* __restrict__ bo, float* __restrict__ out) {
  __shared__ float wos[kC * kO], bos[kO];
  int tid = threadIdx.x;
  if (tid < kC * kO) wos[tid] = wo[tid];
  if (tid < kO) bos[tid] = bo[tid];
  __syncthreads();
  int p = blockIdx.x * 256 + tid;
  const float* zr = z + (size_t)p * kC;
  float zv[kC];
#pragma unroll
  for (int c = 0; c < kC; ++c) zv[c] = fmaxf(zr[c], 0.0f);
#pragma unroll
  for (int o = 0; o < kO; ++o) {
    float a = bos[o];
#pragma unroll
    for (int c = 0; c < kC; ++c) a += zv[c] * wos[c * kO + o];
    out[(size_t)p * kO + o] = a;
  }
}

// ================================================================== launch
extern "C" void kernel_launch(void* const* d_in, const int* in_sizes, int n_in,
                              void* d_out, int out_size, void* d_ws, size_t ws_size,
                              hipStream_t stream) {
  (void)in_sizes; (void)n_in; (void)out_size; (void)ws_size;
  const float* x   = (const float*)d_in[0];
  const float* w1  = (const float*)d_in[1];
  const float* b1  = (const float*)d_in[2];
  const float* w2  = (const float*)d_in[3];
  const float* b2  = (const float*)d_in[4];
  const float* fw1 = (const float*)d_in[5];
  const float* fb1 = (const float*)d_in[6];
  const float* gw1 = (const float*)d_in[7];
  const float* gb1 = (const float*)d_in[8];
  const float* fw2 = (const float*)d_in[9];
  const float* fb2 = (const float*)d_in[10];
  const float* gw2 = (const float*)d_in[11];
  const float* gb2 = (const float*)d_in[12];
  const float* fw3 = (const float*)d_in[13];
  const float* fb3 = (const float*)d_in[14];
  const float* gw3 = (const float*)d_in[15];
  const float* gb3 = (const float*)d_in[16];
  const float* wo  = (const float*)d_in[17];
  const float* bo  = (const float*)d_in[18];

  float* out_main = (float*)d_out;                       // (B,N,O)
  float* adj      = out_main + (size_t)kB * kN * kO;     // (B,N,N), returned output #2

  char* ws = (char*)d_ws;
  size_t off = 0;
  auto alloc = [&](size_t bytes) -> void* {
    void* p = ws + off;
    off = (off + bytes + 255) & ~(size_t)255;
    return p;
  };
  _Float16* m1h = (_Float16*)alloc((size_t)kB * kN * kH * sizeof(_Float16));
  _Float16* m2h = (_Float16*)alloc((size_t)kB * kN * kH * sizeof(_Float16));
  float* act1   = (float*)alloc((size_t)kB * kN * 7 * kC * sizeof(float));
  float* hbuf   = (float*)alloc((size_t)kB * kN * kC * sizeof(float));
  float* z0     = (float*)alloc((size_t)kB * kN * kC * sizeof(float));
  float* z1     = (float*)alloc((size_t)kB * kN * kC * sizeof(float));
  float* dis    = (float*)alloc((size_t)kB * kN * sizeof(float));

  embed_kernel<<<kB * kN / 256, 256, 0, stream>>>(x, w1, b1, w2, b2, m1h, m2h);
  adjgemm_kernel<<<kB * kPairs, 128, 0, stream>>>(m1h, m2h, adj);
  softmax_kernel<<<kB * kN, 256, 0, stream>>>(adj, dis);
  conv1_kernel<<<(kB * kN) / 8, 256, 0, stream>>>(x, fw1, fb1, gw1, gb1, act1);
  conv23_kernel<<<(kB * kN) / 8, 256, 0, stream>>>(act1, fw2, fb2, gw2, gb2,
                                                   fw3, fb3, gw3, gb3, hbuf);
  const float* zin = hbuf;
  float* zout = z0;
  for (int it = 0; it < kKprop; ++it) {
    prop_kernel<<<kB * (kN / 64), 128, 0, stream>>>(adj, dis, hbuf, zin, zout);
    zin = zout;
    zout = (zout == z0) ? z1 : z0;
  }
  out_kernel<<<kB * kN / 256, 256, 0, stream>>>(zin, wo, bo, out_main);
}
